// HashingSymbol_42245298323614
// MI455X (gfx1250) — compile-verified, matrix operand order checked by reading, not run on hardware
//
#include <hip/hip_runtime.h>

#define ROWS 4096
#define IND  512
#define KD   256
#define KDP  264          // padded LDS row stride (elements): 528B = 132 dwords -> bank shift 4
#define VD   512
#define NK   8192
#define NS   8
#define TK   32
#define NEGINF (-3.0e38f)

typedef __attribute__((ext_vector_type(16))) __bf16 v16bf;
typedef __attribute__((ext_vector_type(8)))  __bf16 v8bf;
typedef __attribute__((ext_vector_type(8)))  float  v8f;

union V16 { v16bf v; v8bf h[2]; };

__device__ __forceinline__ unsigned short f2bf(float f) {
    unsigned int u = __float_as_uint(f);
    u = (u + 0x7fffu + ((u >> 16) & 1u)) >> 16;
    return (unsigned short)u;
}

// CDNA5 async copy: per-lane 16B global -> LDS, tracked by ASYNCcnt.
__device__ __forceinline__ void async_b128(unsigned ldsaddr, const void* gaddr) {
    asm volatile("global_load_async_to_lds_b128 %0, %1, off"
                 :: "v"(ldsaddr), "v"(gaddr) : "memory");
}
__device__ __forceinline__ void wait_async0() {
    asm volatile("s_wait_asynccnt 0x0" ::: "memory");
}

// ---------------------------------------------------------------------------
// Kernel A: normalize key rows, convert to bf16.  One wave per key row.
// ---------------------------------------------------------------------------
__global__ void norm_keys_kernel(const float* __restrict__ keys,
                                 unsigned short* __restrict__ keysbf) {
    const int wid  = blockIdx.x * (blockDim.x >> 5) + (threadIdx.x >> 5);
    const int lane = threadIdx.x & 31;
    const float* kr = keys + (size_t)wid * KD;
    float4 a = *(const float4*)(kr + lane * 8);
    float4 b = *(const float4*)(kr + lane * 8 + 4);
    float ss = a.x*a.x + a.y*a.y + a.z*a.z + a.w*a.w
             + b.x*b.x + b.y*b.y + b.z*b.z + b.w*b.w;
    #pragma unroll
    for (int m = 16; m >= 1; m >>= 1) ss += __shfl_xor(ss, m, 32);
    const float inv = rsqrtf(ss);
    uint4 p;
    p.x = (unsigned)f2bf(a.x*inv) | ((unsigned)f2bf(a.y*inv) << 16);
    p.y = (unsigned)f2bf(a.z*inv) | ((unsigned)f2bf(a.w*inv) << 16);
    p.z = (unsigned)f2bf(b.x*inv) | ((unsigned)f2bf(b.y*inv) << 16);
    p.w = (unsigned)f2bf(b.z*inv) | ((unsigned)f2bf(b.w*inv) << 16);
    *(uint4*)(keysbf + (size_t)wid * KD + lane * 8) = p;
}

// ---------------------------------------------------------------------------
// Kernel B: q = x @ wq^T + bq (f32) + bf16 copy.  One block per row.
// ---------------------------------------------------------------------------
__global__ void qproj_kernel(const float* __restrict__ x,
                             const float* __restrict__ wq,
                             const float* __restrict__ bq,
                             float* __restrict__ qf,
                             unsigned short* __restrict__ qbf) {
    const int row = blockIdx.x;
    const int tid = threadIdx.x;           // 256
    __shared__ float xr[IND];
    xr[tid]       = x[(size_t)row * IND + tid];
    xr[tid + 256] = x[(size_t)row * IND + tid + 256];
    __syncthreads();
    const float* wrow = wq + (size_t)tid * IND;
    float acc = bq[tid];
    #pragma unroll 8
    for (int k = 0; k < IND; ++k) acc = fmaf(xr[k], wrow[k], acc);
    qf [(size_t)row * KD + tid] = acc;
    qbf[(size_t)row * KD + tid] = f2bf(acc);
}

// ---------------------------------------------------------------------------
// Kernel R: resid = q @ wr^T + br (f32).  One block per row.
// ---------------------------------------------------------------------------
__global__ void resid_kernel(const float* __restrict__ qf,
                             const float* __restrict__ wr,
                             const float* __restrict__ br,
                             float* __restrict__ resid) {
    const int row = blockIdx.x;
    const int tid = threadIdx.x;           // 256
    __shared__ float qr[KD];
    qr[tid] = qf[(size_t)row * KD + tid];
    __syncthreads();
    for (int c = tid; c < VD; c += 256) {
        const float* w = wr + (size_t)c * KD;
        float acc = br[c];
        #pragma unroll 8
        for (int k = 0; k < KD; ++k) acc = fmaf(qr[k], w[k], acc);
        resid[(size_t)row * VD + c] = acc;
    }
}

// ---------------------------------------------------------------------------
// Kernel C: bf16 WMMA score GEMM + streaming exact top-32.
// grid = (64 row-tiles of 64 rows, 8 slots), block = 128 (4 waves).
// Key tiles (16 keys x 256 dims, bf16) are async-copied into double-buffered
// LDS (ASYNCcnt) and shared by all 4 waves; wave w computes scores for q-rows
// rowBase + 16w .. +15 (its own WMMA B operand held in registers).
// A = keys (M=16 x K=32), B = q (K=32 x N=16); D lane = q-row, VGPR r = key.
// Two independent accumulator chains + a scheduling fence let the 16
// ds_load_b128 fragments issue ahead of the WMMA burst (partial dscnt waits).
// ---------------------------------------------------------------------------
__global__ void __launch_bounds__(128)
scores_topk_kernel(const unsigned short* __restrict__ qbf,
                   const unsigned short* __restrict__ keysbf,
                   float* __restrict__ tsc,
                   int*   __restrict__ tix) {
    const int rowBase = blockIdx.x * 64;
    const int slot    = blockIdx.y;
    const int tid     = threadIdx.x;       // 0..127
    const int wave    = tid >> 5;          // 0..3
    const int lane    = tid & 31;
    const int nhalf   = lane & 15;
    const int khalf   = (lane >> 4) ? 8 : 0;

    __shared__ __align__(16) unsigned short kbuf[2][16 * KDP];  // 2 x 8.25KB
    __shared__ float sSc[128][TK];
    __shared__ int   sIx[128][TK];

    // --- preload B: this wave's 16 q-rows, 8 k-steps of 32 dims ---
    const unsigned short* qrow = qbf + (size_t)(rowBase + wave * 16 + nhalf) * KD;
    V16 Breg[8];
    #pragma unroll
    for (int kk = 0; kk < 8; ++kk) {
        const int kb = kk * 32 + khalf;
        Breg[kk].h[0] = *(const v8bf*)(qrow + kb);
        Breg[kk].h[1] = *(const v8bf*)(qrow + kb + 16);
    }

    #pragma unroll
    for (int j = 0; j < TK; ++j) { sSc[tid][j] = NEGINF; sIx[tid][j] = 0; }
    float curMin = NEGINF;
    int   minPos = 0;

    const unsigned short* keyBase = keysbf + (size_t)slot * NK * KD;
    const unsigned lds0 = (unsigned)(size_t)&kbuf[0][0];
    const unsigned lds1 = (unsigned)(size_t)&kbuf[1][0];

    // per-thread copy job: 64B of one key row (8 threads per row)
    const int crow = tid >> 3;
    const int csub = tid & 7;

    // prologue: stage tile 0 into buffer 0
    {
        const unsigned short* g = keyBase + crow * KD + csub * 32;
        const unsigned l = lds0 + crow * (KDP * 2) + csub * 64;
        #pragma unroll
        for (int i = 0; i < 4; ++i) async_b128(l + i * 16, g + i * 8);
    }
    wait_async0();
    __syncthreads();

    for (int t = 0; t < NK / 16; ++t) {
        const int cur = t & 1;
        // issue async copy of tile t+1 into the other buffer (overlaps WMMA)
        if (t + 1 < NK / 16) {
            const unsigned short* g =
                keyBase + (size_t)(t + 1) * 16 * KD + crow * KD + csub * 32;
            const unsigned l = (cur ? lds0 : lds1) + crow * (KDP * 2) + csub * 64;
            #pragma unroll
            for (int i = 0; i < 4; ++i) async_b128(l + i * 16, g + i * 8);
        }

        // consume tile t from LDS: issue all 16 ds_load_b128, then WMMA burst
        const unsigned short* krow = &kbuf[cur][nhalf * KDP];
        V16 A[8];
        #pragma unroll
        for (int kk = 0; kk < 8; ++kk) {
            const int kb = kk * 32 + khalf;
            A[kk].h[0] = *(const v8bf*)(krow + kb);
            A[kk].h[1] = *(const v8bf*)(krow + kb + 16);
        }
        asm volatile("" ::: "memory");   // fence: keep ds_loads ahead of WMMAs

        v8f acc0 = {0.f, 0.f, 0.f, 0.f, 0.f, 0.f, 0.f, 0.f};
        v8f acc1 = {0.f, 0.f, 0.f, 0.f, 0.f, 0.f, 0.f, 0.f};
        #pragma unroll
        for (int kk = 0; kk < 8; kk += 2) {
            acc0 = __builtin_amdgcn_wmma_f32_16x16x32_bf16(
                     false, A[kk].v,     false, Breg[kk].v,     (short)0, acc0, false, false);
            acc1 = __builtin_amdgcn_wmma_f32_16x16x32_bf16(
                     false, A[kk + 1].v, false, Breg[kk + 1].v, (short)0, acc1, false, false);
        }
        const v8f acc = acc0 + acc1;

        // streaming exact per-lane top-32 (key id = t*16 + r + khalf)
        #pragma unroll
        for (int r = 0; r < 8; ++r) {
            const float s = acc[r];
            if (s > curMin) {
                sSc[tid][minPos] = s;
                sIx[tid][minPos] = t * 16 + r + khalf;
                curMin = sSc[tid][0]; minPos = 0;
                #pragma unroll
                for (int j = 1; j < TK; ++j) {
                    const float v = sSc[tid][j];
                    if (v < curMin) { curMin = v; minPos = j; }
                }
            }
        }

        wait_async0();      // tile t+1 landed (this wave's batch)
        __syncthreads();    // all waves: copies landed + done reading tile t
    }

    // --- merge per row: 2 disjoint per-lane lists (64 candidates) -> top-32 ---
    if (tid < 64) {
        const int w = tid >> 4, n = tid & 15;
        const int row = rowBase + tid;
        float* osc = tsc + ((size_t)slot * ROWS + row) * TK;
        int*   oix = tix + ((size_t)slot * ROWS + row) * TK;
        const int ct0 = w * 32 + n, ct1 = ct0 + 16;
        for (int sel = 0; sel < TK; ++sel) {
            float best = NEGINF; int bt = ct0, bj = 0;
            for (int j = 0; j < TK; ++j) {
                const float v0 = sSc[ct0][j];
                if (v0 > best) { best = v0; bt = ct0; bj = j; }
                const float v1 = sSc[ct1][j];
                if (v1 > best) { best = v1; bt = ct1; bj = j; }
            }
            osc[sel] = best;
            oix[sel] = sIx[bt][bj];
            sSc[bt][bj] = NEGINF;
        }
    }
}

// ---------------------------------------------------------------------------
// Kernel D: softmax over top-32, weighted gather of values, add residual.
// grid = (4096 rows, 8 slots), block = 128.  Values table (128MB) is
// L2-resident, so the 32-way gathers hit L2.
// ---------------------------------------------------------------------------
__global__ void gather_out_kernel(const float* __restrict__ vals,
                                  const float* __restrict__ resid,
                                  const float* __restrict__ tsc,
                                  const int*   __restrict__ tix,
                                  float* __restrict__ out) {
    const int row  = blockIdx.x;
    const int slot = blockIdx.y;
    const int tid  = threadIdx.x;          // 128
    __shared__ float sS[TK];
    __shared__ int   sI[TK];
    const size_t tb = ((size_t)slot * ROWS + row) * TK;
    if (tid < TK) { sS[tid] = tsc[tb + tid]; sI[tid] = tix[tb + tid]; }
    __syncthreads();

    float mx = NEGINF;
    #pragma unroll
    for (int j = 0; j < TK; ++j) mx = fmaxf(mx, sS[j]);
    float sum = 0.f;
    float wl[TK]; int il[TK];
    #pragma unroll
    for (int j = 0; j < TK; ++j) { wl[j] = __expf(sS[j] - mx); sum += wl[j]; il[j] = sI[j]; }
    const float inv = 1.0f / sum;
    #pragma unroll
    for (int j = 0; j < TK; ++j) wl[j] *= inv;

    const float* vbase = vals + (size_t)slot * NK * VD;
    for (int d = tid; d < VD; d += 128) {
        float acc = 0.f;
        #pragma unroll
        for (int j = 0; j < TK; ++j)
            acc = fmaf(wl[j], vbase[(size_t)il[j] * VD + d], acc);
        out[((size_t)row * NS + slot) * VD + d] = acc + resid[(size_t)row * VD + d];
    }
}

// ---------------------------------------------------------------------------
extern "C" void kernel_launch(void* const* d_in, const int* in_sizes, int n_in,
                              void* d_out, int out_size, void* d_ws, size_t ws_size,
                              hipStream_t stream) {
    const float* x     = (const float*)d_in[0];   // 8*512*512
    const float* bkeys = (const float*)d_in[1];   // 8*8192*256
    const float* bvals = (const float*)d_in[2];   // 8*8192*512
    const float* wq    = (const float*)d_in[3];   // 256*512
    const float* bq    = (const float*)d_in[4];   // 256
    const float* wr    = (const float*)d_in[5];   // 512*256
    const float* br    = (const float*)d_in[6];   // 512
    float* out = (float*)d_out;                   // 4096*8*512

    unsigned char* ws = (unsigned char*)d_ws;
    unsigned short* keysbf = (unsigned short*)(ws);              // 33,554,432 B
    unsigned short* qbf    = (unsigned short*)(ws + 33554432);   //  2,097,152 B
    float*          qf     = (float*)(ws + 35651584);            //  4,194,304 B
    float*          resid  = (float*)(ws + 39845888);            //  8,388,608 B
    float*          tsc    = (float*)(ws + 48234496);            //  4,194,304 B
    int*            tix    = (int*)  (ws + 52428800);            //  4,194,304 B

    norm_keys_kernel<<<NS * NK / 8, 256, 0, stream>>>(bkeys, keysbf);
    qproj_kernel<<<ROWS, 256, 0, stream>>>(x, wq, bq, qf, qbf);
    resid_kernel<<<ROWS, 256, 0, stream>>>(qf, wr, br, resid);
    scores_topk_kernel<<<dim3(ROWS / 64, NS), 128, 0, stream>>>(qbf, keysbf, tsc, tix);
    gather_out_kernel<<<dim3(ROWS, NS), 128, 0, stream>>>(bvals, resid, tsc, tix, out);
}
